// PsiQRHTransformerComplete_81913616269399
// MI455X (gfx1250) — matrix operand order repaired and checked by reading, not compiled, not run on hardware
//
#include <hip/hip_runtime.h>
#include <hip/hip_bf16.h>
#include <math.h>

typedef _Float16 v16h __attribute__((ext_vector_type(16)));
typedef _Float16 h8v  __attribute__((ext_vector_type(8)));
typedef float    v8f  __attribute__((ext_vector_type(8)));
typedef unsigned int u32x4 __attribute__((ext_vector_type(4)));
typedef int          i32x8 __attribute__((ext_vector_type(8)));
typedef int          i32x4 __attribute__((ext_vector_type(4)));

#define VOCABN   128000
#define BATCH    32
#define TLEN     4096
#define KMIN     41        // first rfft bin with f > 0.01
#define NFREQ    2007      // bins 41..2047 inclusive
#define OMEGA_F  6.28318530717958647692f
#define KWAVE_F  12.5663706143591729539f   // 2*pi / 0.5
#define CHIRP_F  0.01f
#define EPSF     1e-10f

#define XH_BYTES (BATCH * TLEN * 2)        // 256 KB f16 copy of x (fits in one WGP's LDS)

// ---------------------------------------------------------------------------
// Kernel A: one-time f32 -> f16 conversion of x into workspace.
// ---------------------------------------------------------------------------
__global__ __launch_bounds__(256) void cvt_kernel(const float* __restrict__ x,
                                                  _Float16* __restrict__ xh) {
    int i = blockIdx.x * 256 + threadIdx.x;
    if (i < BATCH * TLEN) xh[i] = (_Float16)x[i];
}

// ---------------------------------------------------------------------------
// Kernel 0: constants for the log-log regression (Lmean, var) + zero cov[32].
// ---------------------------------------------------------------------------
__global__ __launch_bounds__(256) void setup_stats_kernel(float* __restrict__ W) {
    __shared__ float red[256];
    const int t = threadIdx.x;

    float s = 0.f;
    for (int i = t; i < NFREQ; i += 256) {
        float f = (float)(KMIN + i) / (float)TLEN;
        s += __logf(f + EPSF);
    }
    red[t] = s; __syncthreads();
    for (int o = 128; o > 0; o >>= 1) { if (t < o) red[t] += red[t + o]; __syncthreads(); }
    float Lmean = red[0] / (float)NFREQ;
    __syncthreads();

    float v = 0.f;
    for (int i = t; i < NFREQ; i += 256) {
        float f = (float)(KMIN + i) / (float)TLEN;
        float d = __logf(f + EPSF) - Lmean;
        v += d * d;
    }
    red[t] = v; __syncthreads();
    for (int o = 128; o > 0; o >>= 1) { if (t < o) red[t] += red[t + o]; __syncthreads(); }

    if (t == 0) { W[0] = Lmean; W[1] = red[0]; }
    if (t < BATCH) W[2 + t] = 0.f;   // cov accumulators (re-zeroed every call)
}

// ---------------------------------------------------------------------------
// Kernel 1: masked DFT power spectrum as a WMMA GEMM.
// Grid 32 x 128 threads (4 waves). TDM pulls the entire 256 KB f16 x image
// into LDS once per workgroup (tensor_load_to_lds + s_wait_tensorcnt), then
// each wave runs an independent 16-frequency GEMM column block:
//   A (16x32 f16): ds_load_b128 from LDS, ISA 7.12.2 layout
//   B (32x16 f16): cos/sin basis via in-register complex rotation
//   D (16x16 f32): VGPR r <-> batch row r + 8*(lane>=16), column = lane&15
// ---------------------------------------------------------------------------
__global__ __launch_bounds__(128) void spectrum_cov_kernel(const _Float16* __restrict__ xh,
                                                           float* __restrict__ W) {
    extern __shared__ _Float16 shx[];               // 256 KB dynamic LDS

    const int lane = threadIdx.x & 31;
    const int wid  = threadIdx.x >> 5;              // wave id 0..3

    // ---- TDM: load the whole [32 x 4096] f16 tensor into LDS (wave 0 only)
    if (wid == 0) {
        const unsigned int lds_off = (unsigned int)(uintptr_t)(void*)shx;   // LDS offset = low 32 bits
        const unsigned long long ga = (unsigned long long)(uintptr_t)xh;

        u32x4 g0;
        g0[0] = 1u;                                             // count=1, user mode, no gather
        g0[1] = lds_off;                                        // lds_addr [63:32]
        g0[2] = (unsigned int)(ga & 0xffffffffu);               // global_addr [95:64]
        g0[3] = (unsigned int)((ga >> 32) & 0x01ffffffu)        // global_addr [120:96]
              | (2u << 30);                                     // type=2 ("image") [127:126]

        i32x8 g1;
        g1[0] = 0x00010000;        // wg_mask=0, data_size=1 (2 bytes), no barrier/iter/pad
        g1[1] = (TLEN & 0xffff) << 16;                          // tensor_dim0[15:0] in [63:48]
        g1[2] = (TLEN >> 16) | (BATCH << 16);                   // tensor_dim0[31:16] | tensor_dim1[15:0]
        g1[3] = (TLEN & 0xffff) << 16;                          // tensor_dim1 hi=0 | tile_dim0=4096
        g1[4] = BATCH;                                          // tile_dim1=32, tile_dim2=0
        g1[5] = TLEN;                                           // tensor_dim0_stride[31:0]
        g1[6] = 0;                                              // stride hi | dim1_stride lo
        g1[7] = 0;

        i32x4 g2 = {0, 0, 0, 0};                                // 2D tensor: groups 2/3 unused
        i32x4 g3 = {0, 0, 0, 0};

#if defined(__clang_major__) && (__clang_major__ >= 23)
        i32x8 g4 = {0, 0, 0, 0, 0, 0, 0, 0};
        __builtin_amdgcn_tensor_load_to_lds(g0, g1, g2, g3, g4, 0);
#else
        __builtin_amdgcn_tensor_load_to_lds(g0, g1, g2, g3, 0);
#endif
        __builtin_amdgcn_s_wait_tensorcnt(0);
    }
    __syncthreads();

    // ---- per-wave frequency block
    const int fb      = blockIdx.x * 4 + wid;       // 0..127 (126,127 fully masked)
    const int n_local = lane & 15;
    const int nIdx    = fb * 16 + n_local;
    const int kfreq   = KMIN + nIdx;
    const float w     = (OMEGA_F / (float)TLEN) * (float)kfreq;
    const int  kbase  = (lane < 16) ? 0 : 8;        // B layout: K chunk offset per lane group

    float c[16], s[16];
#pragma unroll
    for (int j = 0; j < 16; ++j) {
        __sincosf(w * (float)(kbase + j + (j & 8)), &s[j], &c[j]);
    }
    float S32, C32;
    __sincosf(w * 32.0f, &S32, &C32);

    const int arow = lane & 15;
    const int aoff = (lane < 16) ? 0 : 8;           // A layout: K chunk offset per lane group

    v8f acc_re0 = {}, acc_im0 = {}, acc_re1 = {}, acc_im1 = {};

    for (int kb = 0; kb < TLEN; kb += 32) {
        // Pack B tiles (cos basis -> real part, sin basis -> imag part)
        v16h bc, bs;
#pragma unroll
        for (int j = 0; j < 16; ++j) { bc[j] = (_Float16)c[j]; bs[j] = (_Float16)s[j]; }

        // A fragments straight from LDS (16B ds_load_b128 each)
        const h8v* pa0 = (const h8v*)(shx + (size_t)arow        * TLEN + kb + aoff);
        const h8v* pa1 = (const h8v*)(shx + (size_t)(arow + 16) * TLEN + kb + aoff);
        h8v lo0 = pa0[0], hi0 = pa0[2];             // K chunks at +0 and +16 elements
        h8v lo1 = pa1[0], hi1 = pa1[2];
        v16h a0 = __builtin_shufflevector(lo0, hi0, 0,1,2,3,4,5,6,7,8,9,10,11,12,13,14,15);
        v16h a1 = __builtin_shufflevector(lo1, hi1, 0,1,2,3,4,5,6,7,8,9,10,11,12,13,14,15);

        acc_re0 = __builtin_amdgcn_wmma_f32_16x16x32_f16(false, a0, false, bc, (short)0, acc_re0, false, false);
        acc_im0 = __builtin_amdgcn_wmma_f32_16x16x32_f16(false, a0, false, bs, (short)0, acc_im0, false, false);
        acc_re1 = __builtin_amdgcn_wmma_f32_16x16x32_f16(false, a1, false, bc, (short)0, acc_re1, false, false);
        acc_im1 = __builtin_amdgcn_wmma_f32_16x16x32_f16(false, a1, false, bs, (short)0, acc_im1, false, false);

        // Rotate basis forward by 32 samples
#pragma unroll
        for (int j = 0; j < 16; ++j) {
            float nc = c[j] * C32 - s[j] * S32;
            float ns = s[j] * C32 + c[j] * S32;
            c[j] = nc; s[j] = ns;
        }
    }

    // cov_b += (log k - Lmean) * log(P[b,k] + eps), summed over this wave's freqs
    const float Lmean = W[0];
    const float lkc   = __logf((float)kfreq / (float)TLEN + EPSF) - Lmean;
    const bool  valid = (nIdx < NFREQ);

#pragma unroll
    for (int r = 0; r < 8; ++r) {
        float P0 = acc_re0[r] * acc_re0[r] + acc_im0[r] * acc_im0[r];
        float P1 = acc_re1[r] * acc_re1[r] + acc_im1[r] * acc_im1[r];
        float v0 = valid ? lkc * __logf(P0 + EPSF) : 0.f;
        float v1 = valid ? lkc * __logf(P1 + EPSF) : 0.f;
        for (int m = 8; m >= 1; m >>= 1) {
            v0 += __shfl_xor(v0, m, 32);
            v1 += __shfl_xor(v1, m, 32);
        }
        if ((lane & 15) == 0) {
            int b0 = r + 8 * (lane >> 4);            // batches 0..15 (acc*_0)
            atomicAdd(&W[2 + b0], v0);
            atomicAdd(&W[2 + b0 + 16], v1);          // batches 16..31 (acc*_1)
        }
    }
}

// ---------------------------------------------------------------------------
// Kernel 2: finalize alpha[b] from cov and var.
// ---------------------------------------------------------------------------
__global__ __launch_bounds__(32) void alpha_kernel(float* __restrict__ W) {
    int b = threadIdx.x;
    if (b < BATCH) {
        float var  = W[1];
        float cov  = W[2 + b];
        float beta = -cov / (var + EPSF);
        float D    = fminf(fmaxf((3.0f - beta) * 0.5f, 0.5f), 1.5f);
        float a    = 1.0f + 0.8f * (D - 1.0f);
        W[34 + b]  = fminf(fmaxf(a, 0.1f), 3.0f);
    }
}

// ---------------------------------------------------------------------------
// Kernel 3: logits [32, 128000]. HBM-store bound (16.4 MB); coalesced per row.
// ---------------------------------------------------------------------------
__global__ __launch_bounds__(256) void logits_kernel(const float* __restrict__ psi,
                                                     const float* __restrict__ W,
                                                     float* __restrict__ out) {
    __shared__ float sh_alpha[BATCH], sh_p0[BATCH], sh_p1[BATCH];
    const int t = threadIdx.x;
    if (t < BATCH) {
        sh_alpha[t] = W[34 + t];
        sh_p0[t]    = psi[t * 4 + 0];
        sh_p1[t]    = psi[t * 4 + 1];
    }
    __syncthreads();

    const int v = blockIdx.x * 256 + t;
    if (v >= VOCABN) return;

    const float lam = (float)v * (1.0f / (float)VOCABN);
    const float ph  = OMEGA_F - KWAVE_F * lam + CHIRP_F * lam * lam;
    float sph, cph;
    __sincosf(ph, &sph, &cph);

#pragma unroll 4
    for (int b = 0; b < BATCH; ++b) {
        float amp = __sinf(OMEGA_F + sh_alpha[b] * lam);
        float wr  = amp * cph;
        float wi  = amp * sph;
        float nrm = sqrtf(wr * wr + wi * wi);
        float cp  = (wr * sh_p0[b] + wi * sh_p1[b]) / (nrm + 1e-8f);
        out[(size_t)b * VOCABN + v] = fabsf(cp) * 10.0f;   // (coupling^2)^0.5 / 0.1
    }
}

// ---------------------------------------------------------------------------
extern "C" void kernel_launch(void* const* d_in, const int* in_sizes, int n_in,
                              void* d_out, int out_size, void* d_ws, size_t ws_size,
                              hipStream_t stream) {
    const float* x   = (const float*)d_in[0];   // [32, 4096] f32
    const float* psi = (const float*)d_in[1];   // [32, 4]    f32
    float* out = (float*)d_out;                 // [32, 128000] f32

    _Float16* xh = (_Float16*)d_ws;                          // [0 .. 256KB): f16 copy of x
    float*    W  = (float*)((char*)d_ws + XH_BYTES);         // stats: [0]=Lmean [1]=var [2..33]=cov [34..65]=alpha

    cvt_kernel          <<<(BATCH * TLEN + 255) / 256, 256, 0, stream>>>(x, xh);
    setup_stats_kernel  <<<1, 256, 0, stream>>>(W);
    spectrum_cov_kernel <<<32, 128, XH_BYTES, stream>>>(xh, W);
    alpha_kernel        <<<1, 32, 0, stream>>>(W);
    logits_kernel       <<<(VOCABN + 255) / 256, 256, 0, stream>>>(psi, W, out);
}
